// GraphTransformerNet_79439715107320
// MI455X (gfx1250) — matrix operand order repaired and checked by reading, not compiled, not run on hardware
//
#include <hip/hip_runtime.h>
#include <hip/hip_bf16.h>

// GraphTransformerNet forward for MI455X (gfx1250, wave32).
// Dense q/k/v/skip projections -> v_wmma_f32_16x16x32_bf16 (bf16 in, f32 acc),
// 32x128 block tile, 4 waves x (16x64) each = 4 WMMA per 32-deep k-step with
// the A fragment reused across all 4; fragments are pure ds_load_b128 streams.
// Weights are packed ONCE per layer into an N-major (transposed) bf16 buffer
// so both A and B tile staging are 16B global->LDS vector copies (no in-GEMM
// transpose, no scalar u16 gathers).
// Edge phase (memory-bound: ~20+ GB mostly L2-resident at 23.3 TB/s / 192 MB
// L2) keeps channel-coalesced scatter atomics and computes e = edge_attr*We
// on the fly (never materialized: saves 1.6 GB/layer).
//
// Fixed sizes: N=100000 (divisible by 32), E=1600000, K in {128,256}, M=1024.

typedef __attribute__((ext_vector_type(16))) __bf16 v16bf;
typedef __attribute__((ext_vector_type(8)))  __bf16 v8bf;
typedef __attribute__((ext_vector_type(8)))  float  v8f;

#define N_NODES  100000
#define N_EDGES  1600000
#define G_GRAPHS 64
#define D_IN     128
#define DMODEL   256
#define QCOLS    1024   // [q|k|v|s]

// ---- order-preserving float <-> uint for atomicMax-based segment_max ----
static __device__ __forceinline__ unsigned f32_ord(float f) {
  unsigned u = __float_as_uint(f);
  return (u & 0x80000000u) ? ~u : (u | 0x80000000u);
}
static __device__ __forceinline__ float ord_f32(unsigned u) {
  return __uint_as_float((u & 0x80000000u) ? (u & 0x7fffffffu) : ~u);
}
#define ORD_NEG_INF 0x007FFFFFu  // f32_ord(-inf)

// ------------------------------ fills -----------------------------------
__global__ void fill_f32_kernel(float* p, float v, int n) {
  int t = blockIdx.x * blockDim.x + threadIdx.x;
  if (t < n) p[t] = v;
}
__global__ void fill_u32_kernel(unsigned* p, unsigned v, int n) {
  int t = blockIdx.x * blockDim.x + threadIdx.x;
  if (t < n) p[t] = v;
}

// ------------------------- f32 -> bf16 convert --------------------------
__global__ void convert_bf16_kernel(const float* __restrict__ in,
                                    __bf16* __restrict__ out, int n) {
  int t = blockIdx.x * blockDim.x + threadIdx.x;
  if (t < n) out[t] = (__bf16)in[t];
}

// ---- pack [Wq|Wk|Wv|Ws] (each K x 256, row-major) -> bf16 TRANSPOSED ----
// Wcat_t[col][k], col in 0..1023 -> [q|k|v|s] block, so GEMM B staging is a
// contiguous vector copy. Runs once per layer on a <=256K-element matrix.
__global__ void pack_w_kernel(const float* __restrict__ Wq, const float* __restrict__ Wk,
                              const float* __restrict__ Wv, const float* __restrict__ Ws,
                              __bf16* __restrict__ Wcat_t, int K) {
  int t = blockIdx.x * blockDim.x + threadIdx.x;
  if (t >= K * QCOLS) return;
  int col = t / K, k = t - col * K;       // t = col*K + k  (dest contiguous)
  int sel = col >> 8, c = col & 255;
  const float* W = (sel == 0) ? Wq : (sel == 1) ? Wk : (sel == 2) ? Wv : Ws;
  Wcat_t[t] = (__bf16)W[k * DMODEL + c];
}
__global__ void pack_bias_kernel(const float* __restrict__ bq, const float* __restrict__ bk,
                                 const float* __restrict__ bv, const float* __restrict__ bs,
                                 float* __restrict__ bcat) {
  int t = blockIdx.x * blockDim.x + threadIdx.x;
  if (t >= QCOLS) return;
  int sel = t >> 8, c = t & 255;
  const float* b = (sel == 0) ? bq : (sel == 1) ? bk : (sel == 2) ? bv : bs;
  bcat[t] = b[c];
}

// --------------------------- WMMA GEMM ----------------------------------
// C[N x M] = A[N x K](bf16, row-major) @ B(given as Bt[M x K], N-major) + bias.
// Block = 128 threads = 4 waves; 32x128 block tile; waves in 2x2, each wave
// owns a 16x64 strip = 4 WMMA tiles per 32-deep k-step (A frag reused 4x).
// Requires N % 32 == 0, K % 32 == 0, M % 128 == 0 (true here).
#define LDSPAD 40   // bf16 elems/row: 80 B = 20 banks (conflict-free), 16B-aligned
__global__ __launch_bounds__(128)
void gemm_bf16_wmma_kernel(const __bf16* __restrict__ A,
                           const __bf16* __restrict__ Btg,   // [M][K]
                           const float*  __restrict__ bias,
                           float* __restrict__ C,
                           int K, int M) {
  __shared__ __align__(16) __bf16 As[32][LDSPAD];   // A tile, row-major
  __shared__ __align__(16) __bf16 Bt[128][LDSPAD];  // B tile, N-major: Bt[n][k]

  const int tid   = threadIdx.x;        // 0..127
  const int wave  = tid >> 5;
  const int lane  = tid & 31;
  const int group = lane >> 4;          // wave32 lane group 0/1
  const int lrow  = lane & 15;
  const int wr    = (wave >> 1) * 16;   // wave row offset in 32
  const int wc    = (wave & 1) * 64;    // wave col offset in 128
  const int r0    = blockIdx.y * 32;
  const int c0    = blockIdx.x * 128;

  const int ldr = tid >> 2;             // A staging: row 0..31
  const int ldc = (tid & 3) * 8;        // 8 contiguous bf16 (16 B)

  v8f acc[4] = {};
  for (int k0 = 0; k0 < K; k0 += 32) {
    __syncthreads();
    // A tile: one 16B global load -> one 16B LDS store per thread
    *(v8bf*)&As[ldr][ldc] =
        *(const v8bf*)&A[(long long)(r0 + ldr) * K + (k0 + ldc)];
    // B tile: weights are pre-transposed, so each thread copies its whole
    // 32-bf16 (64 B) k-run with four 16B vector load/store pairs. The 0.5 MB
    // weight matrix is L2-resident across all row-blocks.
    {
      const __bf16* bsrc = &Btg[(long long)(c0 + tid) * K + k0];
      *(v8bf*)&Bt[tid][0]  = *(const v8bf*)&bsrc[0];
      *(v8bf*)&Bt[tid][8]  = *(const v8bf*)&bsrc[8];
      *(v8bf*)&Bt[tid][16] = *(const v8bf*)&bsrc[16];
      *(v8bf*)&Bt[tid][24] = *(const v8bf*)&bsrc[24];
    }
    __syncthreads();

    // A fragment (ISA 7.12.2, 16-bit A 16x32): lane row M=lrow;
    //   elems 0..7 -> K = 8*group + i ; elems 8..15 -> K = 16 + 8*group + (i-8)
    // => two contiguous 8-elem runs => two ds_load_b128.
    const v8bf* ar = (const v8bf*)&As[wr + lrow][0];
    v8bf a0 = ar[group], a1 = ar[group + 2];
    v16bf afrag = __builtin_shufflevector(a0, a1,
        0, 1, 2, 3, 4, 5, 6, 7, 8, 9, 10, 11, 12, 13, 14, 15);

#pragma unroll
    for (int s = 0; s < 4; ++s) {
      // B fragment (32x16): lane col N=lrow; elems 0..7 -> K = 8*group + i ;
      // elems 8..15 -> K = 16 + 8*group + (i-8) (same as A) => two b128 loads.
      const v8bf* br = (const v8bf*)&Bt[wc + s * 16 + lrow][0];
      v8bf b0 = br[group], b1 = br[group + 2];
      v16bf bfrag = __builtin_shufflevector(b0, b1,
          0, 1, 2, 3, 4, 5, 6, 7, 8, 9, 10, 11, 12, 13, 14, 15);
      acc[s] = __builtin_amdgcn_wmma_f32_16x16x32_bf16(
          false, afrag, false, bfrag, (short)0, acc[s], false, false);
    }
  }

  // C/D f32 layout: VGPR r -> M = 8*group + r, N = lrow.
#pragma unroll
  for (int s = 0; s < 4; ++s) {
    const int n = c0 + wc + s * 16 + lrow;
    const float bn = bias[n];
#pragma unroll
    for (int r = 0; r < 8; ++r) {
      int m = r0 + wr + group * 8 + r;
      C[(long long)m * M + n] = acc[s][r] + bn;
    }
  }
}

// ---------------- edge phase: attention logits + segment max ------------
// qkvs row layout: [ q(0..255) | k(256..511) | v(512..767) | s(768..1023) ]
__global__ void edge_alpha_kernel(const float* __restrict__ qkvs,
                                  const int* __restrict__ ei,
                                  const float* __restrict__ ea,
                                  const float* __restrict__ We,
                                  float* __restrict__ alpha,
                                  unsigned* __restrict__ amax) {
  int t = blockIdx.x * blockDim.x + threadIdx.x;   // E*4 threads
  if (t >= N_EDGES * 4) return;
  int e = t >> 2, h = t & 3;
  int src = ei[e], dst = ei[N_EDGES + e];
  float a = ea[e];
  // all three bases are 256B-aligned (h*64 floats) -> b128 loads
  const float4* q4 = (const float4*)(qkvs + (long long)dst * QCOLS + h * 64);
  const float4* k4 = (const float4*)(qkvs + (long long)src * QCOLS + 256 + h * 64);
  const float4* w4 = (const float4*)(We + h * 64);
  float acc = 0.f;
#pragma unroll
  for (int i = 0; i < 16; ++i) {
    float4 qv = q4[i], kv = k4[i], wv = w4[i];
    acc += qv.x * (kv.x + a * wv.x) + qv.y * (kv.y + a * wv.y) +
           qv.z * (kv.z + a * wv.z) + qv.w * (kv.w + a * wv.w);
  }
  acc *= 0.125f;                                   // 1/sqrt(64)
  alpha[t] = acc;
  atomicMax(&amax[dst * 4 + h], f32_ord(acc));
}

__global__ void edge_expsum_kernel(const int* __restrict__ ei,
                                   float* __restrict__ alpha,
                                   const unsigned* __restrict__ amax,
                                   float* __restrict__ asum) {
  int t = blockIdx.x * blockDim.x + threadIdx.x;   // E*4 threads
  if (t >= N_EDGES * 4) return;
  int e = t >> 2, h = t & 3;
  int dst = ei[N_EDGES + e];
  float w = expf(alpha[t] - ord_f32(amax[dst * 4 + h]));
  alpha[t] = w;
  atomicAdd(&asum[dst * 4 + h], w);
}

// --------- message scatter: aggr[dst, j] += (v[src,j] + ea*We[j]) * coef ---
// Thread t -> (edge, channel j). Adjacent lanes hit adjacent channels of the
// same dst row -> coalesced atomic traffic within cachelines.
__global__ void edge_scatter_kernel(const float* __restrict__ qkvs,
                                    const int* __restrict__ ei,
                                    const float* __restrict__ ea,
                                    const float* __restrict__ We,
                                    const float* __restrict__ alpha,
                                    const float* __restrict__ asum,
                                    float* __restrict__ aggr) {
  long long t = (long long)blockIdx.x * blockDim.x + threadIdx.x;  // E*256
  if (t >= (long long)N_EDGES * DMODEL) return;
  int e = (int)(t >> 8), j = (int)(t & 255), h = j >> 6;
  int src = ei[e], dst = ei[N_EDGES + e];
  float coef = alpha[e * 4 + h] / (asum[dst * 4 + h] + 1e-16f);
  float msg  = (qkvs[(long long)src * QCOLS + 512 + j] + ea[e] * We[j]) * coef;
  atomicAdd(&aggr[(long long)dst * DMODEL + j], msg);
}

// ---------------- skip connection + ReLU --------------------------------
__global__ void skip_relu_kernel(const float* __restrict__ aggr,
                                 const float* __restrict__ qkvs,
                                 float* __restrict__ hout) {
  int t = blockIdx.x * blockDim.x + threadIdx.x;   // N*256
  if (t >= N_NODES * DMODEL) return;
  int n = t >> 8, j = t & 255;
  float v = aggr[t] + qkvs[(long long)n * QCOLS + 768 + j];  // s block (bias in GEMM)
  hout[t] = v > 0.f ? v : 0.f;
}

// ---------------- pooling + final linear --------------------------------
__global__ void pool_scatter_kernel(const float* __restrict__ h,
                                    const int* __restrict__ batch,
                                    float* __restrict__ pooled) {
  int t = blockIdx.x * blockDim.x + threadIdx.x;   // N*256
  if (t >= N_NODES * DMODEL) return;
  int n = t >> 8, j = t & 255;
  atomicAdd(&pooled[batch[n] * DMODEL + j], h[t]);
}

__global__ void pool_final_kernel(const float* __restrict__ pooled,
                                  const float* __restrict__ Wlin,
                                  const float* __restrict__ blin,
                                  float* __restrict__ out) {
  __shared__ float red[256];
  int g = blockIdx.x, j = threadIdx.x;
  red[j] = pooled[g * DMODEL + j] * Wlin[j];
  __syncthreads();
  for (int s = 128; s > 0; s >>= 1) {
    if (j < s) red[j] += red[j + s];
    __syncthreads();
  }
  if (j == 0) out[g] = red[0] + blin[0];
}

// ------------------------------ driver ----------------------------------
extern "C" void kernel_launch(void* const* d_in, const int* in_sizes, int n_in,
                              void* d_out, int out_size, void* d_ws, size_t ws_size,
                              hipStream_t stream) {
  (void)in_sizes; (void)n_in; (void)out_size; (void)ws_size;

  // setup_inputs() dict order; params flattened as a JAX pytree (sorted keys):
  //   0:x 1:edge_index 2:edge_attr 3:batch 4:Wlin 5:blin
  //   then per layer l (base = 6 + 9*l): We,Wk,Wq,Ws,Wv,bk,bq,bs,bv
  const float* x     = (const float*)d_in[0];
  const int*   ei    = (const int*)d_in[1];      // [2, E] flat (src row, dst row)
  const float* ea    = (const float*)d_in[2];    // [E, 1]
  const int*   batch = (const int*)d_in[3];
  const float* Wlin  = (const float*)d_in[4];
  const float* blin  = (const float*)d_in[5];

  // workspace carve-out (256B aligned); total ~695 MB
  char* p = (char*)d_ws;
  auto carve = [&](size_t bytes) {
    char* r = p;
    p += (bytes + 255) & ~(size_t)255;
    return (void*)r;
  };
  float*    qkvs   = (float*)   carve((size_t)N_NODES * QCOLS  * 4);
  float*    aggr   = (float*)   carve((size_t)N_NODES * DMODEL * 4);
  float*    hcur   = (float*)   carve((size_t)N_NODES * DMODEL * 4);
  __bf16*   hbf    = (__bf16*)  carve((size_t)N_NODES * DMODEL * 2);
  float*    alpha  = (float*)   carve((size_t)N_EDGES * 4 * 4);
  unsigned* amax   = (unsigned*)carve((size_t)N_NODES * 4 * 4);
  float*    asum   = (float*)   carve((size_t)N_NODES * 4 * 4);
  __bf16*   Wcat_t = (__bf16*)  carve((size_t)DMODEL * QCOLS * 2);  // [1024][K]
  float*    bcat   = (float*)   carve((size_t)QCOLS * 4);
  float*    pooled = (float*)   carve((size_t)G_GRAPHS * DMODEL * 4);

  const int B = 256;
  for (int l = 0; l < 3; ++l) {
    const int K = (l == 0) ? D_IN : DMODEL;
    const float* Xin = (l == 0) ? x : hcur;
    const int base = 6 + 9 * l;
    const float* We = (const float*)d_in[base + 0];
    const float* Wk = (const float*)d_in[base + 1];
    const float* Wq = (const float*)d_in[base + 2];
    const float* Ws = (const float*)d_in[base + 3];
    const float* Wv = (const float*)d_in[base + 4];
    const float* bk = (const float*)d_in[base + 5];
    const float* bq = (const float*)d_in[base + 6];
    const float* bs = (const float*)d_in[base + 7];
    const float* bv = (const float*)d_in[base + 8];

    // 1) bf16 activations + packed (transposed) bf16 weights / f32 bias
    int cvt = N_NODES * K;
    convert_bf16_kernel<<<(cvt + B - 1) / B, B, 0, stream>>>(Xin, hbf, cvt);
    pack_w_kernel<<<(K * QCOLS + B - 1) / B, B, 0, stream>>>(Wq, Wk, Wv, Ws, Wcat_t, K);
    pack_bias_kernel<<<(QCOLS + B - 1) / B, B, 0, stream>>>(bq, bk, bv, bs, bcat);

    // 2) fused projection GEMM: qkvs = hbf @ Wcat + bcat  (N x K x 1024)
    dim3 gg(QCOLS / 128, N_NODES / 32);
    gemm_bf16_wmma_kernel<<<gg, 128, 0, stream>>>(hbf, Wcat_t, bcat, qkvs, K, QCOLS);

    // 3) scatter-softmax state init
    fill_u32_kernel<<<(N_NODES * 4 + B - 1) / B, B, 0, stream>>>(amax, ORD_NEG_INF, N_NODES * 4);
    fill_f32_kernel<<<(N_NODES * 4 + B - 1) / B, B, 0, stream>>>(asum, 0.f, N_NODES * 4);
    fill_f32_kernel<<<(N_NODES * DMODEL + B - 1) / B, B, 0, stream>>>(aggr, 0.f, N_NODES * DMODEL);

    // 4) edge phase
    edge_alpha_kernel<<<(N_EDGES * 4 + B - 1) / B, B, 0, stream>>>(qkvs, ei, ea, We, alpha, amax);
    edge_expsum_kernel<<<(N_EDGES * 4 + B - 1) / B, B, 0, stream>>>(ei, alpha, amax, asum);
    long long sc = (long long)N_EDGES * DMODEL;
    edge_scatter_kernel<<<(unsigned)((sc + B - 1) / B), B, 0, stream>>>(qkvs, ei, ea, We, alpha, asum, aggr);

    // 5) skip + ReLU -> hcur
    skip_relu_kernel<<<(N_NODES * DMODEL + B - 1) / B, B, 0, stream>>>(aggr, qkvs, hcur);
  }

  // pooling + final linear
  fill_f32_kernel<<<(G_GRAPHS * DMODEL + B - 1) / B, B, 0, stream>>>(pooled, 0.f, G_GRAPHS * DMODEL);
  pool_scatter_kernel<<<(N_NODES * DMODEL + B - 1) / B, B, 0, stream>>>(hcur, batch, pooled);
  pool_final_kernel<<<G_GRAPHS, DMODEL, 0, stream>>>(pooled, Wlin, blin, (float*)d_out);
}
